// SNNLeaky_37151467110512
// MI455X (gfx1250) — matrix operand order repaired and checked by reading, not compiled, not run on hardware
//
#include <hip/hip_runtime.h>

// ---------------------------------------------------------------------------
// Fused SNN-Leaky forward for MI455X (gfx1250, wave32, WMMA).
//   cur[t,b,o] = sum_i x[t,b,i]*W[o,i] + bias[o]   (f16 WMMA, f32 accum)
//   mem = 0.9*mem + cur; spk = (mem>=1); mem -= spk (scan over t in-register)
//
// Key idea vs. previous round: W is t-invariant, so each wave hoists ALL of
// its B-matrix WMMA fragments (2 o-tiles x 16 K-steps x 8 VGPRs = 256 VGPRs)
// into registers once, using gfx1250's 1024-VGPR wave state. The t-loop then
// only streams x through a small LDS staging buffer (f32->f16, coalesced).
// ---------------------------------------------------------------------------

typedef _Float16 v16h __attribute__((ext_vector_type(16)));
typedef _Float16 h8   __attribute__((ext_vector_type(8)));
typedef _Float16 h4   __attribute__((ext_vector_type(4)));
typedef float    v8f  __attribute__((ext_vector_type(8)));

#define T_STEPS 100
#define BDIM    512
#define IDIM    512
#define ODIM    512
#define BT      16              // b rows per workgroup (WMMA M)
#define OT      128             // o cols per workgroup (4 waves x 32)
#define WCOLS   32              // o cols per wave (2 wmma tiles)
#define LPAD    8               // pad (halves) per LDS row -> +16B, banks spread
#define XROW    (IDIM + LPAD)
#define KSTEPS  (IDIM / 32)     // 16

// Load 16 consecutive f32 and convert to a packed v16h B/A fragment register.
__device__ __forceinline__ v16h cvt16(const float4* __restrict__ p) {
    float4 q0 = p[0], q1 = p[1], q2 = p[2], q3 = p[3];
    v16h r = { (_Float16)q0.x, (_Float16)q0.y, (_Float16)q0.z, (_Float16)q0.w,
               (_Float16)q1.x, (_Float16)q1.y, (_Float16)q1.z, (_Float16)q1.w,
               (_Float16)q2.x, (_Float16)q2.y, (_Float16)q2.z, (_Float16)q2.w,
               (_Float16)q3.x, (_Float16)q3.y, (_Float16)q3.z, (_Float16)q3.w };
    return r;
}

__global__ __launch_bounds__(128)
void snn_leaky_fused(const float* __restrict__ x,
                     const float* __restrict__ wgt,
                     const float* __restrict__ bias,
                     float* __restrict__ spk)
{
    __shared__ _Float16 Xl[BT * XROW];   // ~16.6 KB (only x is staged)

    const int tid    = threadIdx.x;
    const int lane   = tid & 31;
    const int wv     = tid >> 5;
    const int b_base = blockIdx.x * BT;
    const int o_base = blockIdx.y * OT;

    // ---- per-lane WMMA fragment coordinates (ISA 7.12.2 layouts) ----
    const int m    = lane & 15;               // A: row M
    const int kbA  = (lane & 16) ? 8 : 0;     // 0 or 8 per lane half (used as mrow)
    const int kbB  = (lane & 16) ? 16 : 0;    // B: K base within 32-step
    const int n    = lane & 15;               // B/C/D: column N
    const int mrow = kbA;                     // C/D: row offset (0 or 8)

    const int oc0 = o_base + wv * WCOLS;      // global col base, tile 0
    const int oc1 = oc0 + 16;                 // tile 1

    const float bias0 = bias[oc0 + n];
    const float bias1 = bias[oc1 + n];

    // ---- hoist ALL B fragments (t-invariant W) into VGPRs: 256 VGPRs ----
    v16h Bf0[KSTEPS], Bf1[KSTEPS];
#pragma unroll
    for (int ks = 0; ks < KSTEPS; ++ks) {
        const float* w0 = wgt + (size_t)(oc0 + n) * IDIM + ks * 32 + kbB;
        const float* w1 = wgt + (size_t)(oc1 + n) * IDIM + ks * 32 + kbB;
        Bf0[ks] = cvt16(reinterpret_cast<const float4*>(w0));
        Bf1[ks] = cvt16(reinterpret_cast<const float4*>(w1));
    }

    float mem0[8], mem1[8];
#pragma unroll
    for (int j = 0; j < 8; ++j) { mem0[j] = 0.f; mem1[j] = 0.f; }

    for (int t = 0; t < T_STEPS; ++t) {
        __syncthreads();   // previous iteration's Xl fragment reads complete
        // ---- stage x[t, b_base:b_base+16, :] f32 -> f16 (coalesced) ----
        {
            const float4* Xg = reinterpret_cast<const float4*>(
                x + ((size_t)t * BDIM + b_base) * IDIM);
            const int total4 = BT * IDIM / 4;             // 2048
            for (int i = tid; i < total4; i += 128) {
                float4 v = Xg[i];
                const int r = i / (IDIM / 4);
                const int c = (i % (IDIM / 4)) * 4;
                h4 h = { (_Float16)v.x, (_Float16)v.y, (_Float16)v.z, (_Float16)v.w };
                *reinterpret_cast<h4*>(&Xl[r * XROW + c]) = h;
            }
        }
        // prefetch next t's x slab (32 KB spread over 128 lanes)
        if (t + 1 < T_STEPS) {
            const char* nx = reinterpret_cast<const char*>(
                x + ((size_t)(t + 1) * BDIM + b_base) * IDIM);
            __builtin_prefetch(nx + tid * 256, 0, 1);
        }
        __syncthreads();

        // ---- seed accumulators with bias ----
        v8f c0, c1;
#pragma unroll
        for (int j = 0; j < 8; ++j) { c0[j] = bias0; c1[j] = bias1; }

        // ---- K loop: only A fragments come from LDS; B lives in VGPRs ----
#pragma unroll
        for (int ks = 0; ks < KSTEPS; ++ks) {
            const _Float16* ap = &Xl[m * XROW + ks * 32 + kbB];
            h8 alo = *reinterpret_cast<const h8*>(ap);
            h8 ahi = *reinterpret_cast<const h8*>(ap + 8);
            v16h a = __builtin_shufflevector(alo, ahi,
                0, 1, 2, 3, 4, 5, 6, 7, 8, 9, 10, 11, 12, 13, 14, 15);

            c0 = __builtin_amdgcn_wmma_f32_16x16x32_f16(
                     false, a, false, Bf0[ks], (short)0, c0, false, false);
            c1 = __builtin_amdgcn_wmma_f32_16x16x32_f16(
                     false, a, false, Bf1[ks], (short)0, c1, false, false);
        }

        // ---- leaky integrate, threshold, reset, store spikes ----
        const size_t out_base = ((size_t)t * BDIM + b_base) * ODIM;
#pragma unroll
        for (int j = 0; j < 8; ++j) {
            const size_t row = out_base + (size_t)(j + mrow) * ODIM;

            float mm = 0.9f * mem0[j] + c0[j];
            float s  = (mm >= 1.0f) ? 1.0f : 0.0f;
            mem0[j]  = mm - s;
            spk[row + (oc0 + n)] = s;

            mm = 0.9f * mem1[j] + c1[j];
            s  = (mm >= 1.0f) ? 1.0f : 0.0f;
            mem1[j]  = mm - s;
            spk[row + (oc1 + n)] = s;
        }
    }
}

extern "C" void kernel_launch(void* const* d_in, const int* in_sizes, int n_in,
                              void* d_out, int out_size, void* d_ws, size_t ws_size,
                              hipStream_t stream) {
    const float* x    = (const float*)d_in[0];   // [T, B, I]
    const float* wgt  = (const float*)d_in[1];   // [O, I]
    const float* bias = (const float*)d_in[2];   // [O]
    float* spk        = (float*)d_out;           // [T, B, O]
    (void)in_sizes; (void)n_in; (void)out_size; (void)d_ws; (void)ws_size;

    dim3 grid(BDIM / BT, ODIM / OT);             // (32, 4)
    snn_leaky_fused<<<grid, 128, 0, stream>>>(x, wgt, bias, spk);
}
